// ResidualBlockNoBNDynamic_12360915878241
// MI455X (gfx1250) — matrix operand order repaired and checked by reading, not compiled, not run on hardware
//
#include <hip/hip_runtime.h>
#include <hip/hip_bf16.h>

// ---------------- CDNA5 vector types ----------------
typedef __attribute__((ext_vector_type(16))) __bf16 v16bf;
typedef __attribute__((ext_vector_type(8)))  __bf16 v8bf;
typedef __attribute__((ext_vector_type(8)))  float  v8f;

// Problem constants
#define NB   16
#define NC   64
#define NH   192
#define NW   192
#define NEXP 5
#define KTOT 576          // 9 taps * 64 in-channels
#define NKC  18           // 576 / 32 K-chunks
// Conv-kernel tiling: 64 out-ch x (8 rows x 16 cols) per block
#define TILE_H 8
#define TILE_W 16
#define IN_H   10
#define IN_W   18
#define CH_STRIDE 72      // 64 channels padded to 72 (144B rows -> 16B aligned b128 gathers)

static __device__ __forceinline__ unsigned short f32_to_bf16(float f) {
  union { float f; unsigned int u; } c; c.f = f;
  unsigned int u = c.u;
  u += 0x7FFFu + ((u >> 16) & 1u);   // round-to-nearest-even
  return (unsigned short)(u >> 16);
}

// ------------------------------------------------------------------
// Kernel 1: aggregate expert weights per sample, emit bf16 directly in
// the CDNA5 16-bit A-matrix (16x32) fragment layout:
//   e<8 : K = half*8 + e ;  e>=8 : K = 16 + half*8 + (e-8) ; lane = half*16 + m
// frag[ (((b*18+kc)*4 + mTile)*32 + lane)*16 + e ]
// ------------------------------------------------------------------
__global__ __launch_bounds__(256)
void dynconv_agg_frag(const float* __restrict__ attn,
                      const float* __restrict__ W1,
                      const float* __restrict__ W2,
                      unsigned short* __restrict__ frag1,
                      unsigned short* __restrict__ frag2) {
  int gid = blockIdx.x * 256 + threadIdx.x;          // 2 * 16 * 64 * 576 threads
  int layer = gid / (NB * NC * KTOT);
  int rem   = gid % (NB * NC * KTOT);
  int b  = rem / (NC * KTOT);
  int r2 = rem % (NC * KTOT);
  int o  = r2 / KTOT;
  int kk = r2 % KTOT;                                // K = tap*64 + in_ch
  int tap = kk >> 6;
  int ic  = kk & 63;

  const float* Wsrc = layer ? W2 : W1;
  float s = 0.f;
#pragma unroll
  for (int e = 0; e < NEXP; ++e) {
    size_t wi = ((size_t)(e * NC + o) * NC + ic) * 9 + tap; // [K,O,I,3,3]
    s += attn[b * NEXP + e] * Wsrc[wi];
  }

  // fragment coordinates
  int kc = kk >> 5;
  int kl = kk & 31;
  int half, el;
  if (kl < 16) { half = kl >> 3; el = kl & 7; }
  else         { int t = kl - 16; half = t >> 3; el = 8 + (t & 7); }
  int mTile = o >> 4;
  int lane  = half * 16 + (o & 15);
  size_t fi = ((((size_t)b * NKC + kc) * 4 + mTile) * 32 + lane) * 16 + el;
  (layer ? frag2 : frag1)[fi] = f32_to_bf16(s);
}

// Kernel 1b: aggregate biases (tiny)
__global__ __launch_bounds__(256)
void dynconv_agg_bias(const float* __restrict__ attn,
                      const float* __restrict__ b1,
                      const float* __restrict__ b2,
                      float* __restrict__ ob1, float* __restrict__ ob2) {
  int gid = blockIdx.x * 256 + threadIdx.x;          // 2 * 16 * 64
  if (gid >= 2 * NB * NC) return;
  int layer = gid / (NB * NC);
  int rem   = gid % (NB * NC);
  int b = rem / NC, o = rem % NC;
  const float* src = layer ? b2 : b1;
  float s = 0.f;
#pragma unroll
  for (int e = 0; e < NEXP; ++e) s += attn[b * NEXP + e] * src[e * NC + o];
  (layer ? ob2 : ob1)[rem] = s;
}

// ------------------------------------------------------------------
// Kernel 2/3: implicit-GEMM 3x3 conv via v_wmma_f32_16x16x32_bf16.
// Block: 256 threads = 8 waves. Tile: 64 out-ch x (8 rows x 16 cols).
// wave w owns output row nt=w and ALL 4 M-tiles: one B-fragment gather
// feeds 4 WMMAs (4x v8f accumulators per wave).
// ------------------------------------------------------------------
template <bool FIRST>
__global__ __launch_bounds__(256)
void dynconv_wmma(const void* __restrict__ src,              // fp32 x (FIRST) / bf16 y1
                  const unsigned short* __restrict__ frag,   // pre-laid-out A fragments
                  const float* __restrict__ bias,
                  const float* __restrict__ identity,        // x (layer 2 only)
                  void* __restrict__ dst) {                  // bf16 y1 / fp32 out
  __shared__ alignas(32) unsigned short lds_in[IN_H * IN_W * CH_STRIDE]; // 25920 B

  const int b  = blockIdx.z;
  const int y0 = blockIdx.y * TILE_H;
  const int x0 = blockIdx.x * TILE_W;

  // ---- stage input halo tile to LDS, channel-innermost, bf16 ----
  // Each thread owns one (row,col) halo position (180 total) and loops over
  // the 64 channels with a constant global stride (no per-element divides).
  {
    int pos = threadIdx.x;
    if (pos < IN_H * IN_W) {
      const int row = pos / IN_W;
      const int col = pos - row * IN_W;
      const int gy = y0 - 1 + row, gx = x0 - 1 + col;
      unsigned short* lp = &lds_in[(row * IN_W + col) * CH_STRIDE];
      if (gy >= 0 && gy < NH && gx >= 0 && gx < NW) {
        size_t gi = ((size_t)b * NC * NH + gy) * NW + gx;   // channel 0
#pragma unroll 8
        for (int ch = 0; ch < NC; ++ch) {
          unsigned short v;
          if (FIRST) v = f32_to_bf16(((const float*)src)[gi + (size_t)ch * NH * NW]);
          else       v = ((const unsigned short*)src)[gi + (size_t)ch * NH * NW];
          lp[ch] = v;
        }
      } else {
#pragma unroll 8
        for (int ch = 0; ch < NC; ++ch) lp[ch] = 0;
      }
    }
  }
  __syncthreads();

  const int lane = threadIdx.x & 31;
  const int nt   = threadIdx.x >> 5;          // output row within tile (0..7)
  const int half = lane >> 4;
  const int px   = lane & 15;

  const v16bf* fragv = (const v16bf*)frag;    // 32-byte per-lane slots

  v8f acc[4];
#pragma unroll
  for (int mt = 0; mt < 4; ++mt) acc[mt] = (v8f){};

  for (int kc = 0; kc < NKC; ++kc) {
    const int tap = kc >> 1, g = kc & 1;
    const int ky = tap / 3, kx = tap - ky * 3;

    if (kc + 1 < NKC) {  // stream next A-fragment chunk -> global_prefetch_b8
      size_t nf = (((size_t)b * NKC + kc + 1) * 4) * 32 + lane;
      __builtin_prefetch(&fragv[nf], 0, 1);
    }

    // B fragment: 32x16 bf16 layout — e<8: K = half*8 + e ; e>=8: K = 16 + half*8 + (e-8),
    // K == input channel within the 32-channel chunk in LDS.
    const int lb = ((nt + ky) * IN_W + (px + kx)) * CH_STRIDE + g * 32 + half * 8;
    v8bf blo = *(const v8bf*)&lds_in[lb];
    v8bf bhi = *(const v8bf*)&lds_in[lb + 16];
    v16bf bf;
#pragma unroll
    for (int i = 0; i < 8; ++i) { bf[i] = blo[i]; bf[i + 8] = bhi[i]; }

    // A fragments (pre-arranged): one 32B contiguous load per lane per tile
    const size_t fb = (((size_t)b * NKC + kc) * 4) * 32 + lane;
#pragma unroll
    for (int mt = 0; mt < 4; ++mt) {
      v16bf a = fragv[fb + (size_t)mt * 32];
      acc[mt] = __builtin_amdgcn_wmma_f32_16x16x32_bf16(false, a, false, bf,
                                                        (short)0, acc[mt], false, false);
    }
  }

  // ---- epilogue: C/D layout — VGPR r, lane l => M = r + 8*half, N = l&15 ----
  const int y  = y0 + nt;
  const int xg = x0 + px;
#pragma unroll
  for (int mt = 0; mt < 4; ++mt) {
#pragma unroll
    for (int r = 0; r < 8; ++r) {
      const int o = mt * 16 + r + half * 8;
      float v = acc[mt][r] + bias[b * NC + o];
      size_t oi = ((size_t)(b * NC + o) * NH + y) * NW + xg;
      if (FIRST) ((unsigned short*)dst)[oi] = f32_to_bf16(fmaxf(v, 0.f));
      else       ((float*)dst)[oi] = identity[oi] + v;
    }
  }
}

// ------------------------------------------------------------------
extern "C" void kernel_launch(void* const* d_in, const int* in_sizes, int n_in,
                              void* d_out, int out_size, void* d_ws, size_t ws_size,
                              hipStream_t stream) {
  (void)in_sizes; (void)n_in; (void)out_size; (void)ws_size;
  const float* x    = (const float*)d_in[0];
  const float* attn = (const float*)d_in[1];
  const float* W1   = (const float*)d_in[2];
  const float* b1   = (const float*)d_in[3];
  const float* W2   = (const float*)d_in[4];
  const float* b2   = (const float*)d_in[5];
  float* out = (float*)d_out;

  // workspace layout
  const size_t FRAG_BYTES = (size_t)NB * NKC * 4 * 32 * 16 * 2; // 1,179,648
  char* ws = (char*)d_ws;
  unsigned short* frag1 = (unsigned short*)(ws);
  unsigned short* frag2 = (unsigned short*)(ws + FRAG_BYTES);
  float* bias1 = (float*)(ws + 2 * FRAG_BYTES);
  float* bias2 = (float*)(ws + 2 * FRAG_BYTES + 4096);
  unsigned short* y1 = (unsigned short*)(ws + 2 * FRAG_BYTES + 8192); // bf16 intermediate

  // 1) per-sample expert aggregation into WMMA fragment layout
  {
    int total = 2 * NB * NC * KTOT;            // 1,179,648
    dynconv_agg_frag<<<total / 256, 256, 0, stream>>>(attn, W1, W2, frag1, frag2);
    dynconv_agg_bias<<<(2 * NB * NC + 255) / 256, 256, 0, stream>>>(attn, b1, b2, bias1, bias2);
  }

  // 2) conv1 + ReLU -> bf16 intermediate ; 3) conv2 + residual -> fp32 out
  dim3 grid(NW / TILE_W, NH / TILE_H, NB);     // 12 x 24 x 16
  dynconv_wmma<true ><<<grid, 256, 0, stream>>>((const void*)x,  frag1, bias1, nullptr, (void*)y1);
  dynconv_wmma<false><<<grid, 256, 0, stream>>>((const void*)y1, frag2, bias2, x,      (void*)out);
}
